// DynEdge_global_var_19335942766752
// MI455X (gfx1250) — compile-verified, matrix operand order checked by reading, not run on hardware
//
#include <hip/hip_runtime.h>

typedef __attribute__((ext_vector_type(16))) _Float16 v16h;
typedef __attribute__((ext_vector_type(8)))  float    v8f;
typedef __attribute__((ext_vector_type(2)))  float    v2f;

namespace {

constexpr int T_NODES = 16384;
constexpr int B_GR    = 64;
constexpr int N_PG    = 256;
constexpr int K_NN    = 8;
constexpr int E_TOT   = T_NODES * K_NN;       // 131072 edges
constexpr int EPG     = N_PG * K_NN;          // 2048 edges/graph
constexpr int F_IN    = 6;
constexpr int HID     = 336;                  // 21 tiles of 16
constexpr int HIDP    = 352;                  // K-padded to 11*32
constexpr int OUTD    = 256;
constexpr int XC_D    = 1030;
constexpr int XC_DP   = 1056;                 // 33*32
constexpr int CH_NODES = 2048;
constexpr int CH_EDGES = CH_NODES * K_NN;     // 16384 rows/chunk
constexpr int NCHUNK   = T_NODES / CH_NODES;  // 8
constexpr float NEG_SLOPE = 0.01f;

__device__ __forceinline__ float leaky(float v) { return v > 0.f ? v : NEG_SLOPE * v; }

// K-offset of the r-th packed f16 pair for a lane (hi = lane>>4) in the
// 16-bit A (16x32, lane&15 = row) / B (32x16, lane&15 = col) WMMA layouts.
__device__ __forceinline__ int kpair(int r, int hi) {
    return ((r & 4) << 2) + (hi << 3) + ((r & 3) << 1);
}

// Fragment whose per-lane data is K-contiguous in memory (A rows, or rows of a
// pre-transposed weight WT[n][k]).  Pairs are contiguous -> merges into b64 loads.
__device__ __forceinline__ v16h frag_rowmajor(const _Float16* __restrict__ row, int k0, int hi) {
    v16h f;
#pragma unroll
    for (int r = 0; r < 8; ++r) {
        const int k = k0 + kpair(r, hi);
        f[2 * r]     = row[k];
        f[2 * r + 1] = row[k + 1];
    }
    return f;
}

// Same, but element-wise difference of two rows (the x_j - x_i half of an edge).
__device__ __forceinline__ v16h frag_rowdiff(const _Float16* __restrict__ rj,
                                             const _Float16* __restrict__ ri,
                                             int k0, int hi) {
    v16h f;
#pragma unroll
    for (int r = 0; r < 8; ++r) {
        const int k = k0 + kpair(r, hi);
        f[2 * r]     = (_Float16)(rj[k] - ri[k]);
        f[2 * r + 1] = (_Float16)(rj[k + 1] - ri[k + 1]);
    }
    return f;
}

// ---------------------------------------------------------------- utility kernels

// w [kin, ncols] (row-major, K-major)  ->  wt [ncols, kpad] fp16 (N-major,
// K-contiguous, zero-padded K).  This makes WMMA B-fragment loads contiguous.
__global__ __launch_bounds__(256)
void k_cvt_w_t(const float* __restrict__ w, _Float16* __restrict__ wt,
               int kin, int kpad, int ncols, int total) {
    const int i = blockIdx.x * 256 + threadIdx.x;
    if (i >= total) return;
    const int n = i / kpad, k = i % kpad;
    wt[i] = (k < kin) ? (_Float16)w[(size_t)k * ncols + n] : (_Float16)0.f;
}

__global__ __launch_bounds__(256)
void k_cvt_act_h(const float* __restrict__ a, _Float16* __restrict__ ah,
                 int din, int strideOut, int total) {
    const int i = blockIdx.x * 256 + threadIdx.x;
    if (i >= total) return;
    const int r = i / strideOut, c = i % strideOut;
    ah[i] = (c < din) ? (_Float16)a[(size_t)r * din + c] : (_Float16)0.f;
}

__global__ __launch_bounds__(256)
void k_rowsq(const float* __restrict__ a, float* __restrict__ sq, int d) {
    const int i = blockIdx.x * 256 + threadIdx.x;
    const float* r = a + (size_t)i * d;
    float s = 0.f;
    for (int k = 0; k < d; ++k) s += r[k] * r[k];
    sq[i] = s;
}

// homophily (4 feats) + per-graph mean of x -> global_x [B,10]
__global__ __launch_bounds__(256)
void k_stats(const float* __restrict__ x, const int* __restrict__ ei,
             float* __restrict__ gx) {
    __shared__ float red[256];
    const int g = blockIdx.x, t = threadIdx.x;
    const int* src = ei;
    const int* dst = ei + E_TOT;
    float acc[10];
#pragma unroll
    for (int c = 0; c < 10; ++c) acc[c] = 0.f;
#pragma unroll
    for (int d = 0; d < F_IN; ++d) acc[d] = x[(size_t)(g * N_PG + t) * F_IN + d];
    for (int e8 = 0; e8 < 8; ++e8) {
        const int e = g * EPG + t * 8 + e8;
        const int s = src[e], d = dst[e];
#pragma unroll
        for (int fi = 0; fi < 4; ++fi)
            acc[6 + fi] += (x[(size_t)s * F_IN + fi] == x[(size_t)d * F_IN + fi]) ? 1.f : 0.f;
    }
    for (int c = 0; c < 10; ++c) {
        red[t] = acc[c];
        __syncthreads();
        for (int o = 128; o > 0; o >>= 1) {
            if (t < o) red[t] += red[t + o];
            __syncthreads();
        }
        if (t == 0) gx[g * 10 + c] = (c < 6) ? red[0] / (float)N_PG : red[0] / (float)EPG;
        __syncthreads();
    }
}

// kNN for layer 1 (D=6): one graph per block, all features in LDS.
__global__ __launch_bounds__(256)
void k_knn6(const float* __restrict__ x, int* __restrict__ idx) {
    __shared__ float f[N_PG][F_IN];
    __shared__ float sq[N_PG];
    const int g = blockIdx.x, i = threadIdx.x;
    float mine[F_IN];
    float s = 0.f;
#pragma unroll
    for (int d = 0; d < F_IN; ++d) {
        const float v = x[(size_t)(g * N_PG + i) * F_IN + d];
        mine[d] = v; f[i][d] = v; s += v * v;
    }
    sq[i] = s;
    __syncthreads();
    float bd[K_NN]; int bi[K_NN];
#pragma unroll
    for (int k = 0; k < K_NN; ++k) { bd[k] = 3.4e38f; bi[k] = 0; }
    for (int j = 0; j < N_PG; ++j) {
        float dot = 0.f;
#pragma unroll
        for (int d = 0; d < F_IN; ++d) dot += mine[d] * f[j][d];
        const float dist = s + sq[j] - 2.f * dot;
        if (dist < bd[K_NN - 1]) {           // strict < keeps earlier index on ties (stable top_k)
            int p = K_NN - 1;
            while (p > 0 && dist < bd[p - 1]) { bd[p] = bd[p - 1]; bi[p] = bi[p - 1]; --p; }
            bd[p] = dist; bi[p] = j;
        }
    }
#pragma unroll
    for (int k = 0; k < K_NN; ++k)
        idx[(size_t)(g * N_PG + i) * K_NN + k] = g * N_PG + bi[k];
}

// Per-graph distance matrix via exact fp32 WMMA Gram: dist = sq_i + sq_j - 2*G.
__global__ __launch_bounds__(256)
void k_gram_dist(const float* __restrict__ H, const float* __restrict__ sq,
                 float* __restrict__ dist) {
    const int lane = threadIdx.x & 31;
    const int wid  = blockIdx.x * 8 + (threadIdx.x >> 5);
    const int g  = wid >> 8;
    const int t  = wid & 255;
    const int ti = t >> 4, tj = t & 15;
    const int m  = lane & 15, hi = lane >> 4;
    const float* rowi = H + (size_t)(g * N_PG + ti * 16 + m) * OUTD;
    const float* rowj = H + (size_t)(g * N_PG + tj * 16 + m) * OUTD;
    v8f c = {};
#if __has_builtin(__builtin_amdgcn_wmma_f32_16x16x4_f32)
#pragma unroll 8
    for (int k0 = 0; k0 < OUTD; k0 += 4) {
        v2f a, b;
        a[0] = rowi[k0 + 2 * hi]; a[1] = rowi[k0 + 1 + 2 * hi];   // A 16x4 f32 layout
        b[0] = rowj[k0 + 2 * hi]; b[1] = rowj[k0 + 1 + 2 * hi];   // B 4x16 (sym.)
        c = __builtin_amdgcn_wmma_f32_16x16x4_f32(false, a, false, b, (short)0, c, false, false);
    }
#else
#pragma unroll
    for (int r = 0; r < 8; ++r) {
        const float* ri = H + (size_t)(g * N_PG + ti * 16 + r + 8 * hi) * OUTD;
        const float* rj = H + (size_t)(g * N_PG + tj * 16 + m) * OUTD;
        float s2 = 0.f;
        for (int d = 0; d < OUTD; ++d) s2 += ri[d] * rj[d];
        c[r] = s2;
    }
#endif
    const int j  = tj * 16 + m;
    const float sj = sq[g * N_PG + j];
    float* dg = dist + (size_t)g * N_PG * N_PG;
#pragma unroll
    for (int r = 0; r < 8; ++r) {
        const int i = ti * 16 + r + 8 * hi;
        dg[(size_t)i * N_PG + j] = sq[g * N_PG + i] + sj - 2.f * c[r];
    }
}

__global__ __launch_bounds__(256)
void k_topk8(const float* __restrict__ dist, int* __restrict__ idx) {
    const int g = blockIdx.x, i = threadIdx.x;
    const float* row = dist + ((size_t)g * N_PG + i) * N_PG;
    float bd[K_NN]; int bi[K_NN];
#pragma unroll
    for (int k = 0; k < K_NN; ++k) { bd[k] = 3.4e38f; bi[k] = 0; }
    for (int j = 0; j < N_PG; ++j) {
        const float d = row[j];
        if (d < bd[K_NN - 1]) {
            int p = K_NN - 1;
            while (p > 0 && d < bd[p - 1]) { bd[p] = bd[p - 1]; bi[p] = bi[p - 1]; --p; }
            bd[p] = d; bi[p] = j;
        }
    }
#pragma unroll
    for (int k = 0; k < K_NN; ++k)
        idx[(size_t)(g * N_PG + i) * K_NN + k] = g * N_PG + bi[k];
}

// ---------------------------------------------------------------- WMMA GEMM kernels
// All weight matrices below are pre-transposed fp16: WT[n][k], K-contiguous,
// so B-fragments load with the same contiguous-pair pattern as A-fragments.

// GEMM1 of EdgeConv: A rows are edges, gathered on the fly as [x_i, x_j - x_i].
// hidden = leaky(A @ W1 + b1)  (fp16 out, [CH_EDGES, HIDP])
__global__ __launch_bounds__(256)
void k_gemm_edge_hidden(const _Float16* __restrict__ Hh, int strideH, int D,
                        const int* __restrict__ idx,
                        const _Float16* __restrict__ W1T, int ldk,
                        const float* __restrict__ b1,
                        _Float16* __restrict__ hid,
                        int nodeBase, int ksteps) {
    const int lane = threadIdx.x & 31;
    const int wid  = blockIdx.x * 8 + (threadIdx.x >> 5);
    const int tn = wid % (HID / 16);
    const int tm = wid / (HID / 16);
    const int m  = lane & 15, hi = lane >> 4;
    const int le = tm * 16 + m;                       // local edge row (A row for this lane)
    const int ni = nodeBase + (le >> 3);
    const int nj = idx[(size_t)ni * K_NN + (le & 7)];
    const _Float16* pi = Hh + (size_t)ni * strideH;
    const _Float16* pj = Hh + (size_t)nj * strideH;
    const int col = tn * 16 + m;
    const _Float16* wrow = W1T + (size_t)col * ldk;
    v8f c = {};
    for (int ks = 0; ks < ksteps; ++ks) {
        const int k0 = ks * 32;
        v16h a;
        if (k0 + 32 <= D) {                           // entirely in the x_i half
            a = frag_rowmajor(pi, k0, hi);
        } else if (k0 >= D && k0 + 32 <= 2 * D) {     // entirely in the (x_j - x_i) half
            a = frag_rowdiff(pj - D, pi - D, k0, hi); // index k maps to k-D
        } else {                                      // layer-1 padded step (D=6)
#pragma unroll
            for (int r = 0; r < 8; ++r) {
#pragma unroll
                for (int e = 0; e < 2; ++e) {
                    const int k = k0 + kpair(r, hi) + e;
                    _Float16 v = (_Float16)0.f;
                    if (k < D)          v = pi[k];
                    else if (k < 2 * D) v = (_Float16)(pj[k - D] - pi[k - D]);
                    a[2 * r + e] = v;
                }
            }
        }
        const v16h b = frag_rowmajor(wrow, k0, hi);
        c = __builtin_amdgcn_wmma_f32_16x16x32_f16(false, a, false, b, (short)0, c, false, false);
    }
    const float bias = b1[col];
#pragma unroll
    for (int r = 0; r < 8; ++r) {
        const int row = tm * 16 + r + 8 * hi;
        hid[(size_t)row * HIDP + col] = (_Float16)leaky(c[r] + bias);
    }
}

// GEMM2 of EdgeConv with fused max over K=8 edges: a 16-row output tile is
// exactly 2 nodes; per-lane max over its 8 accumulator VGPRs == max_k.
__global__ __launch_bounds__(256)
void k_gemm_hidden_max(const _Float16* __restrict__ hid,
                       const _Float16* __restrict__ W2T,   // [OUTD, HIDP]
                       const float* __restrict__ b2,
                       float* __restrict__ out, int nodeBase) {
    const int lane = threadIdx.x & 31;
    const int wid  = blockIdx.x * 8 + (threadIdx.x >> 5);
    const int tn = wid & 15, tm = wid >> 4;
    const int m  = lane & 15, hi = lane >> 4;
    const _Float16* arow = hid + (size_t)(tm * 16 + m) * HIDP;
    const int col = tn * 16 + m;
    const _Float16* wrow = W2T + (size_t)col * HIDP;
    v8f c = {};
#pragma unroll
    for (int ks = 0; ks < HIDP / 32; ++ks) {
        const v16h a = frag_rowmajor(arow, ks * 32, hi);
        const v16h b = frag_rowmajor(wrow, ks * 32, hi);
        c = __builtin_amdgcn_wmma_f32_16x16x32_f16(false, a, false, b, (short)0, c, false, false);
    }
    float mx = c[0];
#pragma unroll
    for (int r = 1; r < 8; ++r) mx = fmaxf(mx, c[r]);
    const int node = nodeBase + tm * 2 + hi;
    out[(size_t)node * OUTD + col] = leaky(mx + b2[col]);  // leaky monotone => commutes with max
}

// Plain row-major GEMM + bias + leaky, fp16 output (m1 first layer).
__global__ __launch_bounds__(256)
void k_gemm_rm_h(const _Float16* __restrict__ A, int lda, int ksteps,
                 const _Float16* __restrict__ WT, int ldk,
                 const float* __restrict__ bias,
                 _Float16* __restrict__ outH, int ldo, int nTiles) {
    const int lane = threadIdx.x & 31;
    const int wid  = blockIdx.x * 8 + (threadIdx.x >> 5);
    const int tn = wid % nTiles, tm = wid / nTiles;
    const int m  = lane & 15, hi = lane >> 4;
    const _Float16* arow = A + (size_t)(tm * 16 + m) * lda;
    const int col = tn * 16 + m;
    const _Float16* wrow = WT + (size_t)col * ldk;
    v8f c = {};
    for (int ks = 0; ks < ksteps; ++ks) {
        const v16h a = frag_rowmajor(arow, ks * 32, hi);
        const v16h b = frag_rowmajor(wrow, ks * 32, hi);
        c = __builtin_amdgcn_wmma_f32_16x16x32_f16(false, a, false, b, (short)0, c, false, false);
    }
    const float bb = bias[col];
#pragma unroll
    for (int r = 0; r < 8; ++r) {
        const int row = tm * 16 + r + 8 * hi;
        outH[(size_t)row * ldo + col] = (_Float16)leaky(c[r] + bb);
    }
}

// Plain row-major GEMM + bias + leaky, fp32 output (m1 second layer).
__global__ __launch_bounds__(256)
void k_gemm_rm_f(const _Float16* __restrict__ A, int lda, int ksteps,
                 const _Float16* __restrict__ WT, int ldk,
                 const float* __restrict__ bias,
                 float* __restrict__ outF, int ldo, int nTiles) {
    const int lane = threadIdx.x & 31;
    const int wid  = blockIdx.x * 8 + (threadIdx.x >> 5);
    const int tn = wid % nTiles, tm = wid / nTiles;
    const int m  = lane & 15, hi = lane >> 4;
    const _Float16* arow = A + (size_t)(tm * 16 + m) * lda;
    const int col = tn * 16 + m;
    const _Float16* wrow = WT + (size_t)col * ldk;
    v8f c = {};
    for (int ks = 0; ks < ksteps; ++ks) {
        const v16h a = frag_rowmajor(arow, ks * 32, hi);
        const v16h b = frag_rowmajor(wrow, ks * 32, hi);
        c = __builtin_amdgcn_wmma_f32_16x16x32_f16(false, a, false, b, (short)0, c, false, false);
    }
    const float bb = bias[col];
#pragma unroll
    for (int r = 0; r < 8; ++r) {
        const int row = tm * 16 + r + 8 * hi;
        outF[(size_t)row * ldo + col] = leaky(c[r] + bb);
    }
}

// ---------------------------------------------------------------- tail kernels

__global__ __launch_bounds__(256)
void k_build_xc(const float* __restrict__ x,
                const float* __restrict__ x1, const float* __restrict__ x2,
                const float* __restrict__ x3, const float* __restrict__ x4,
                _Float16* __restrict__ xc, int total) {
    const int i = blockIdx.x * 256 + threadIdx.x;
    if (i >= total) return;
    const int n = i / XC_DP, c = i % XC_DP;
    float v;
    if (c < 6)          v = x[(size_t)n * F_IN + c];
    else if (c < 262)   v = x1[(size_t)n * OUTD + (c - 6)];
    else if (c < 518)   v = x2[(size_t)n * OUTD + (c - 262)];
    else if (c < 774)   v = x3[(size_t)n * OUTD + (c - 518)];
    else if (c < 1030)  v = x4[(size_t)n * OUTD + (c - 774)];
    else                v = 0.f;
    xc[i] = (_Float16)v;
}

__global__ __launch_bounds__(256)
void k_pool(const float* __restrict__ h, float* __restrict__ pooled) {
    const int g = blockIdx.x, c = threadIdx.x;
    float s = 0.f;
    for (int n = 0; n < N_PG; ++n) s += h[(size_t)(g * N_PG + n) * OUTD + c];
    pooled[g * OUTD + c] = s / (float)N_PG;
}

__global__ __launch_bounds__(128)
void k_final(const float* __restrict__ gx, const float* __restrict__ pooled,
             const float* __restrict__ w1, const float* __restrict__ b1,
             const float* __restrict__ w2, const float* __restrict__ b2,
             float* __restrict__ out) {
    __shared__ float z[266];
    __shared__ float hcache[128];
    const int g = blockIdx.x, t = threadIdx.x;
    for (int c = t; c < 266; c += 128)
        z[c] = (c < 10) ? gx[g * 10 + c] : pooled[g * OUTD + (c - 10)];
    __syncthreads();
    float acc = b1[t];
    for (int i = 0; i < 266; ++i) acc += z[i] * w1[i * 128 + t];
    hcache[t] = leaky(acc);
    __syncthreads();
    if (t < 3) {
        float o = b2[t];
        for (int i = 0; i < 128; ++i) o += hcache[i] * w2[i * 3 + t];
        out[g * 3 + t] = o;   // final layer has no activation
    }
}

} // namespace

extern "C" void kernel_launch(void* const* d_in, const int* in_sizes, int n_in,
                              void* d_out, int out_size, void* d_ws, size_t ws_size,
                              hipStream_t stream) {
    (void)in_sizes; (void)n_in; (void)out_size; (void)ws_size;
    const float* x  = (const float*)d_in[0];
    const int*   ei = (const int*)d_in[1];
    const float *w1[4], *b1[4], *w2[4], *b2[4];
    for (int l = 0; l < 4; ++l) {
        w1[l] = (const float*)d_in[3 + 4 * l + 0];
        b1[l] = (const float*)d_in[3 + 4 * l + 1];
        w2[l] = (const float*)d_in[3 + 4 * l + 2];
        b2[l] = (const float*)d_in[3 + 4 * l + 3];
    }
    const float* m1w1 = (const float*)d_in[19];
    const float* m1b1 = (const float*)d_in[20];
    const float* m1w2 = (const float*)d_in[21];
    const float* m1b2 = (const float*)d_in[22];
    const float* m2w1 = (const float*)d_in[23];
    const float* m2b1 = (const float*)d_in[24];
    const float* m2w2 = (const float*)d_in[25];
    const float* m2b2 = (const float*)d_in[26];
    float* out = (float*)d_out;

    // ---- workspace layout (assumed ws_size >= ~160 MB) ----
    char* ws = (char*)d_ws;
    size_t off = 0;
    auto alloc = [&](size_t bytes) -> void* {
        void* p = ws + off;
        off = (off + bytes + 255) & ~(size_t)255;
        return p;
    };
    float* X[4];
    for (int l = 0; l < 4; ++l) X[l] = (float*)alloc((size_t)T_NODES * OUTD * 4);
    float* HOUT = (float*)alloc((size_t)T_NODES * OUTD * 4);
    float* DIST = (float*)alloc((size_t)B_GR * N_PG * N_PG * 4);
    float* SQ   = (float*)alloc((size_t)T_NODES * 4);
    int*   IDX  = (int*)alloc((size_t)T_NODES * K_NN * 4);
    _Float16* HH   = (_Float16*)alloc((size_t)T_NODES * OUTD * 2);
    _Float16* HIDB = (_Float16*)alloc((size_t)CH_EDGES * HIDP * 2);   // also reused by m1 (16384 rows)
    _Float16* XCH  = (_Float16*)alloc((size_t)T_NODES * XC_DP * 2);
    float* POOL = (float*)alloc((size_t)B_GR * OUTD * 4);
    float* GX   = (float*)alloc((size_t)B_GR * 10 * 4);
    // transposed fp16 weights: WT[n][kpad]
    _Float16* W1T[4]; _Float16* W2T[4];
    W1T[0] = (_Float16*)alloc((size_t)HID * 32 * 2);
    for (int l = 1; l < 4; ++l) W1T[l] = (_Float16*)alloc((size_t)HID * 512 * 2);
    for (int l = 0; l < 4; ++l) W2T[l] = (_Float16*)alloc((size_t)OUTD * HIDP * 2);
    _Float16* M1W1T = (_Float16*)alloc((size_t)HID * XC_DP * 2);
    _Float16* M1W2T = (_Float16*)alloc((size_t)OUTD * HIDP * 2);

    // ---- convert weights to transposed/padded fp16 ----
    auto cvtw = [&](const float* w, _Float16* wt, int kin, int kpad, int ncols) {
        const int total = ncols * kpad;
        k_cvt_w_t<<<(total + 255) / 256, 256, 0, stream>>>(w, wt, kin, kpad, ncols, total);
    };
    cvtw(w1[0], W1T[0], 2 * F_IN, 32, HID);
    for (int l = 1; l < 4; ++l) cvtw(w1[l], W1T[l], 2 * OUTD, 2 * OUTD, HID);
    for (int l = 0; l < 4; ++l) cvtw(w2[l], W2T[l], HID, HIDP, OUTD);
    cvtw(m1w1, M1W1T, XC_D, XC_DP, HID);
    cvtw(m1w2, M1W2T, HID, HIDP, OUTD);

    // ---- per-graph stats ----
    k_stats<<<B_GR, 256, 0, stream>>>(x, ei, GX);

    // ---- 4 DynamicEdgeConv layers ----
    const float* lin[4] = { x, X[0], X[1], X[2] };
    for (int l = 0; l < 4; ++l) {
        const int D       = (l == 0) ? F_IN : OUTD;
        const int strideH = (l == 0) ? 8 : OUTD;
        const int ksteps  = (l == 0) ? 1 : (2 * OUTD) / 32;
        const int ldk     = (l == 0) ? 32 : 2 * OUTD;
        if (l == 0) {
            k_knn6<<<B_GR, 256, 0, stream>>>(x, IDX);
        } else {
            k_rowsq<<<T_NODES / 256, 256, 0, stream>>>(lin[l], SQ, OUTD);
            k_gram_dist<<<(B_GR * 256) / 8, 256, 0, stream>>>(lin[l], SQ, DIST);
            k_topk8<<<B_GR, 256, 0, stream>>>(DIST, IDX);
        }
        {
            const int total = T_NODES * strideH;
            k_cvt_act_h<<<(total + 255) / 256, 256, 0, stream>>>(lin[l], HH, D, strideH, total);
        }
        for (int c = 0; c < NCHUNK; ++c) {
            k_gemm_edge_hidden<<<((CH_EDGES / 16) * (HID / 16)) / 8, 256, 0, stream>>>(
                HH, strideH, D, IDX, W1T[l], ldk, b1[l], HIDB, c * CH_NODES, ksteps);
            k_gemm_hidden_max<<<((CH_EDGES / 16) * (OUTD / 16)) / 8, 256, 0, stream>>>(
                HIDB, W2T[l], b2[l], X[l], c * CH_NODES);
        }
    }

    // ---- node MLP (m1) on xc = [x, x1..x4] ----
    {
        const int total = T_NODES * XC_DP;
        k_build_xc<<<(total + 255) / 256, 256, 0, stream>>>(x, X[0], X[1], X[2], X[3], XCH, total);
    }
    k_gemm_rm_h<<<((T_NODES / 16) * (HID / 16)) / 8, 256, 0, stream>>>(
        XCH, XC_DP, XC_DP / 32, M1W1T, XC_DP, m1b1, HIDB, HIDP, HID / 16);
    k_gemm_rm_f<<<((T_NODES / 16) * (OUTD / 16)) / 8, 256, 0, stream>>>(
        HIDB, HIDP, HIDP / 32, M1W2T, HIDP, m1b2, HOUT, OUTD, OUTD / 16);

    // ---- pool + head ----
    k_pool<<<B_GR, 256, 0, stream>>>(HOUT, POOL);
    k_final<<<B_GR, 128, 0, stream>>>(GX, POOL, m2w1, m2b1, m2w2, m2b2, out);
}
